// MyConvolution_77936476553752
// MI455X (gfx1250) — matrix operand order repaired and checked by reading, not compiled, not run on hardware
//
#include <hip/hip_runtime.h>

typedef _Float16 h8  __attribute__((ext_vector_type(8)));
typedef _Float16 h16 __attribute__((ext_vector_type(16)));
typedef float    v8f __attribute__((ext_vector_type(8)));

#define C_IN   64
#define C_OUT  128
#define HW     112
#define KSZ    576          // 64 * 3 * 3
#define ROWPAD 584          // LDS A row stride in halves (584/2=292 dwords, 292%64=36=4*odd -> distinct start banks)
#define SMEM_BYTES (64 * ROWPAD * 2)   // 74752 B dynamic LDS (As, reused as Cs)

#define SHUF16(lo, hi) __builtin_shufflevector((lo), (hi), 0,1,2,3,4,5,6,7,8,9,10,11,12,13,14,15)

// Kernel 0: convert fp32 OIHW weights to fp16 in scratch (K = c*9+r*3+s is contiguous per o).
__global__ void conv_w2h(const float* __restrict__ w, _Float16* __restrict__ wh, int n) {
    int i = blockIdx.x * blockDim.x + threadIdx.x;
    if (i < n) wh[i] = (_Float16)w[i];
}

// Main kernel: implicit-GEMM 3x3 conv via v_wmma_f32_16x16x32_f16.
// Block = 256 threads = 8 waves; block tile = 64 pixels (4 h x 16 w) x 128 channels.
// Wave wv owns channels [wv*16, wv*16+16) and all 4 M-subtiles.
__global__ __launch_bounds__(256)
void conv3x3_wmma(const float* __restrict__ x,
                  const _Float16* __restrict__ wh,
                  const float* __restrict__ bias,
                  float* __restrict__ out) {
    extern __shared__ char smem[];
    _Float16* As = (_Float16*)smem;   // [64][ROWPAD] fp16 im2col tile
    float*    Cs = (float*)smem;      // reused after compute: [8 waves][4][16][16] f32

    const int tid = threadIdx.x;
    const int bx  = blockIdx.x;
    const int wt  = bx % 7;
    const int ht  = (bx / 7) % 28;
    const int n   = bx / 196;
    const int h0  = ht * 4;
    const int w0  = wt * 16;

    // ---- Stage A: im2col fp32 -> fp16 into LDS. e = k*64 + m so consecutive
    // threads hit consecutive w (coalesced 64B global reads; input is L2-resident).
    for (int e = tid; e < KSZ * 64; e += 256) {
        const int m  = e & 63;         // pixel in tile: lh*16 + lw
        const int k  = e >> 6;         // c*9 + r*3 + s
        const int lw = m & 15;
        const int lh = m >> 4;
        const int c  = k / 9;
        const int rs = k - c * 9;
        const int r  = rs / 3;
        const int s  = rs - r * 3;
        const int hi = h0 + lh + r - 1;
        const int wi = w0 + lw + s - 1;
        float v = 0.0f;
        if ((unsigned)hi < HW && (unsigned)wi < HW)
            v = x[((n * C_IN + c) * HW + hi) * HW + wi];
        As[m * ROWPAD + k] = (_Float16)v;
    }
    __syncthreads();

    const int wv    = tid >> 5;
    const int lane  = tid & 31;
    const int half  = lane >> 4;       // 0: lanes 0-15, 1: lanes 16-31
    const int mlo   = lane & 15;
    const int chb   = wv * 16;

    // B fragment source: one fp16 weight row per lane (channel chb+mlo), K contiguous.
    const _Float16* wrow = wh + (chb + mlo) * KSZ;
    __builtin_prefetch(wrow, 0, 0);    // global_prefetch_b8

    // A fragment sources: row M = lane&15 of each 16-pixel subtile.
    const _Float16* arow[4];
    #pragma unroll
    for (int t = 0; t < 4; ++t) arow[t] = As + (t * 16 + mlo) * ROWPAD;

    const int koffA = half * 8;        // A lane half: K {koffA..koffA+7} and {16+koffA..23+koffA}
    const int koffB = half * 8;        // B lane half: K {koffB..koffB+7} and {16+koffB..23+koffB}

    v8f acc[4] = {};

    #pragma unroll 2
    for (int k0 = 0; k0 < KSZ; k0 += 32) {
        // Issue ALL loads up front so the 4 WMMAs can run back-to-back
        // behind a single (partial) dscnt wait instead of load->wait->wmma chains.
        const h8 b0 = *(const h8*)(wrow + k0 + koffB);
        const h8 b1 = *(const h8*)(wrow + k0 + koffB + 16);
        h16 afrag[4];
        #pragma unroll
        for (int t = 0; t < 4; ++t) {
            const h8 a0 = *(const h8*)(arow[t] + k0 + koffA);        // ds_load_b128
            const h8 a1 = *(const h8*)(arow[t] + k0 + koffA + 16);   // ds_load_b128
            afrag[t] = SHUF16(a0, a1);
        }
        const h16 bfrag = SHUF16(b0, b1);
        #pragma unroll
        for (int t = 0; t < 4; ++t)
            acc[t] = __builtin_amdgcn_wmma_f32_16x16x32_f16(
                false, afrag[t], false, bfrag, (short)0, acc[t], false, false);
    }

    __syncthreads();   // everyone done reading As; reuse LDS as Cs

    // Write accumulators: C/D layout -> lane = channel (N), VGPR r = pixel row (M), +8 for upper half.
    float* Csw = Cs + wv * 1024;
    #pragma unroll
    for (int t = 0; t < 4; ++t)
        #pragma unroll
        for (int r = 0; r < 8; ++r)
            Csw[t * 256 + mlo * 16 + (r + half * 8)] = acc[t][r];
    __syncthreads();

    // Coalesced store: lanes 0-15 sweep consecutive w for a channel row; add bias.
    // Output is written once and never re-read -> non-temporal stores (th:TH_NT)
    // so the 205 MB output stream does not evict the L2-resident input/weights.
    #pragma unroll
    for (int t = 0; t < 4; ++t) {
        #pragma unroll
        for (int j = 0; j < 8; ++j) {
            const int chl = j * 2 + half;
            const float v = Csw[t * 256 + chl * 16 + mlo] + bias[chb + chl];
            __builtin_nontemporal_store(
                v, &out[((n * C_OUT + chb + chl) * HW + (h0 + t)) * HW + (w0 + mlo)]);
        }
    }
}

extern "C" void kernel_launch(void* const* d_in, const int* in_sizes, int n_in,
                              void* d_out, int out_size, void* d_ws, size_t ws_size,
                              hipStream_t stream) {
    const float* x  = (const float*)d_in[0];   // (32,64,112,112)
    const float* w  = (const float*)d_in[1];   // (128,64,3,3)
    const float* b  = (const float*)d_in[2];   // (128,)
    float* out      = (float*)d_out;           // (32,128,112,112)
    _Float16* wh    = (_Float16*)d_ws;         // 73728 * 2B = 147456 B scratch

    const int wn = C_OUT * C_IN * 9;           // 73728
    conv_w2h<<<(wn + 255) / 256, 256, 0, stream>>>(w, wh, wn);

    const int nblocks = 32 * 28 * 7;           // n * h-tiles * w-tiles = 6272
    conv3x3_wmma<<<nblocks, 256, SMEM_BYTES, stream>>>(x, wh, b, out);
}
